// GatedDeltaNetBlock_85246510891201
// MI455X (gfx1250) — matrix operand order repaired and checked by reading, not compiled, not run on hardware
//
#include <hip/hip_runtime.h>
#include <hip/hip_bf16.h>

typedef _Float16 h16_t;
typedef __attribute__((ext_vector_type(16))) _Float16 v16h;
typedef __attribute__((ext_vector_type(8)))  float    v8f;

#define RMS_EPS 1e-6f

// ---------------------------------------------------------------------------
// rmsnorm(x, w) -> f16, one block (256 thr) per row
// ---------------------------------------------------------------------------
__global__ void k_rmsnorm_f16(const float* __restrict__ x, const float* __restrict__ w,
                              h16_t* __restrict__ out, int Dd) {
  __shared__ float red[256];
  int row = blockIdx.x, tid = threadIdx.x;
  const float* xr = x + (size_t)row * Dd;
  float s = 0.f;
  for (int i = tid; i < Dd; i += 256) { float v = xr[i]; s += v * v; }
  red[tid] = s; __syncthreads();
  for (int st = 128; st > 0; st >>= 1) {
    if (tid < st) red[tid] += red[tid + st];
    __syncthreads();
  }
  float r = rsqrtf(red[0] / (float)Dd + RMS_EPS);
  for (int i = tid; i < Dd; i += 256)
    out[(size_t)row * Dd + i] = (h16_t)(xr[i] * r * w[i]);
}

// ---------------------------------------------------------------------------
// convert fp32 weight [Kd, N] row-major -> f16 transposed [N, Kd]
// ---------------------------------------------------------------------------
__global__ void k_cvt_tr(const float* __restrict__ in, h16_t* __restrict__ out,
                         int Kd, int N) {
  size_t idx = (size_t)blockIdx.x * 256 + threadIdx.x;
  size_t total = (size_t)Kd * N;
  if (idx >= total) return;
  int k = (int)(idx / N), n = (int)(idx % N);
  out[(size_t)n * Kd + k] = (h16_t)in[idx];
}

// ---------------------------------------------------------------------------
// WMMA GEMM: C[M,N] (f32, + optional residual) = A[M,Kd] (f16) * Bt[N,Kd]^T
// Block = 256 thr (8 waves); block tile 128x64; wave tile 16x64; K-step 32.
// A fragment: lane m = lane&15, two contiguous 16B chunks (K hi*16.., hi*16+8..)
// B fragment: lane n = lane&15, one contiguous 32B chunk (K = hi*16 .. +15)
// C fragment: VGPR r -> (m0 + hi*8 + r, n0 + t*16 + (lane&15))
// ---------------------------------------------------------------------------
__global__ __launch_bounds__(256)
void k_gemm_wmma(const h16_t* __restrict__ A, const h16_t* __restrict__ Bt,
                 float* __restrict__ C, const float* __restrict__ res,
                 int M, int N, int Kd) {
  int wv = threadIdx.x >> 5, lane = threadIdx.x & 31;
  int hi = lane >> 4, lr = lane & 15;
  int m0 = blockIdx.x * 128 + wv * 16;
  int n0 = blockIdx.y * 64;

  const h16_t* pa = A + (size_t)(m0 + lr) * Kd + hi * 16;
  v8f zero = {0.f, 0.f, 0.f, 0.f, 0.f, 0.f, 0.f, 0.f};
  v8f acc[4];
#pragma unroll
  for (int t = 0; t < 4; ++t) acc[t] = zero;

  union { v16h v; int4 q[2]; } a;
  for (int k0 = 0; k0 < Kd; k0 += 32) {
    a.q[0] = *(const int4*)(pa + k0);
    a.q[1] = *(const int4*)(pa + k0 + 8);
    __builtin_prefetch(pa + k0 + 128, 0, 0);
#pragma unroll
    for (int t = 0; t < 4; ++t) {
      v16h b = *(const v16h*)(Bt + (size_t)(n0 + t * 16 + lr) * Kd + k0 + hi * 16);
      acc[t] = __builtin_amdgcn_wmma_f32_16x16x32_f16(
          false, a.v, false, b, (short)0, acc[t], false, false);
    }
  }

#pragma unroll
  for (int t = 0; t < 4; ++t) {
    int n = n0 + t * 16 + lr;
#pragma unroll
    for (int r = 0; r < 8; ++r) {
      int m = m0 + hi * 8 + r;
      size_t idx = (size_t)m * N + n;
      float v = acc[t][r];
      if (res) v += res[idx];
      C[idx] = v;
    }
  }
}

// ---------------------------------------------------------------------------
// beta = sigmoid(h@wb); g = -exp(A_log)*softplus(h@wa + dt_bias)
// one wave (32 thr) per row; threads 0..15 -> beta cols, 16..31 -> g cols
// ---------------------------------------------------------------------------
__global__ void k_small_proj(const h16_t* __restrict__ h16, const float* __restrict__ wb,
                             const float* __restrict__ wa, const float* __restrict__ A_log,
                             const float* __restrict__ dt_bias,
                             float* __restrict__ beta, float* __restrict__ g,
                             int Dd, int Hh) {
  int row = blockIdx.x, tid = threadIdx.x;
  const h16_t* hr = h16 + (size_t)row * Dd;
  if (tid < Hh) {
    int c = tid; float acc = 0.f;
    for (int i = 0; i < Dd; ++i) acc += (float)hr[i] * wb[(size_t)i * Hh + c];
    beta[(size_t)row * Hh + c] = 1.f / (1.f + expf(-acc));
  } else if (tid < 2 * Hh) {
    int c = tid - Hh; float acc = 0.f;
    for (int i = 0; i < Dd; ++i) acc += (float)hr[i] * wa[(size_t)i * Hh + c];
    float xv = acc + dt_bias[c];
    float sp = (xv > 20.f) ? xv : log1pf(expf(xv));
    g[(size_t)row * Hh + c] = -expf(A_log[c]) * sp;
  }
}

// ---------------------------------------------------------------------------
// depthwise causal conv1d (window 4) + silu; one thread per (b,t,c)
// ---------------------------------------------------------------------------
__global__ void k_conv_silu(const float* __restrict__ in, const float* __restrict__ w,
                            float* __restrict__ out, int Cc, int Tt, long total) {
  long idx = (long)blockIdx.x * 256 + threadIdx.x;
  if (idx >= total) return;
  int c = (int)(idx % Cc);
  long bt = idx / Cc;
  int t = (int)(bt % Tt);
  float acc = 0.f;
#pragma unroll
  for (int j = 0; j < 4; ++j) {
    int ts = t - 3 + j;
    if (ts >= 0) acc += in[idx + (long)(j - 3) * Cc] * w[c * 4 + j];
  }
  out[idx] = acc / (1.f + expf(-acc));
}

// ---------------------------------------------------------------------------
// in-place l2norm over last dim (64) with scale; block 64 thr per row
// ---------------------------------------------------------------------------
__global__ void k_l2norm(float* __restrict__ buf, float scale) {
  __shared__ float red[64];
  int row = blockIdx.x, tid = threadIdx.x;
  float* p = buf + (size_t)row * 64;
  float v = p[tid];
  red[tid] = v * v; __syncthreads();
  for (int st = 32; st > 0; st >>= 1) {
    if (tid < st) red[tid] += red[tid + st];
    __syncthreads();
  }
  p[tid] = v * rsqrtf(red[0] + RMS_EPS) * scale;
}

// ---------------------------------------------------------------------------
// gated delta rule scan; one block (128 thr) per (b, head); S[64] in regs/thr
// ---------------------------------------------------------------------------
__global__ __launch_bounds__(128)
void k_scan(const float* __restrict__ qc, const float* __restrict__ kc,
            const float* __restrict__ vc, const float* __restrict__ g,
            const float* __restrict__ beta, float* __restrict__ o,
            int Tt, int Hh) {
  const int DKc = 64, DVc = 128;
  int b = blockIdx.x / Hh, hd = blockIdx.x % Hh;
  int tid = threadIdx.x;
  int Kc = Hh * DKc, Vc = Hh * DVc;
  __shared__ float sk[64], sq[64];
  __shared__ float s_eg, s_b;
  float S[64];
#pragma unroll
  for (int i = 0; i < 64; ++i) S[i] = 0.f;

  for (int t = 0; t < Tt; ++t) {
    size_t rt = (size_t)(b * Tt + t);
    __syncthreads();
    if (tid < 64) sk[tid] = kc[rt * Kc + hd * DKc + tid];
    else          sq[tid - 64] = qc[rt * Kc + hd * DKc + (tid - 64)];
    if (tid == 0) { s_eg = expf(g[rt * Hh + hd]); s_b = beta[rt * Hh + hd]; }
    __syncthreads();
    float eg = s_eg, bt_ = s_b;
    float vt = vc[rt * Vc + hd * DVc + tid];
    float kS = 0.f;
#pragma unroll
    for (int dk = 0; dk < 64; ++dk) { S[dk] *= eg; kS += sk[dk] * S[dk]; }
    float u = (vt - kS) * bt_;
    float ot = 0.f;
#pragma unroll
    for (int dk = 0; dk < 64; ++dk) { S[dk] += sk[dk] * u; ot += sq[dk] * S[dk]; }
    o[rt * Vc + hd * DVc + tid] = ot;
  }
}

// ---------------------------------------------------------------------------
// per-head rmsnorm(o) * gate * sigmoid(gate) -> f16; block 128 per (b,t,h)
// ---------------------------------------------------------------------------
__global__ void k_gate(const float* __restrict__ o, const float* __restrict__ gp,
                       const float* __restrict__ onw, h16_t* __restrict__ og16) {
  __shared__ float red[128];
  int row = blockIdx.x, tid = threadIdx.x;
  float v = o[(size_t)row * 128 + tid];
  red[tid] = v * v; __syncthreads();
  for (int st = 64; st > 0; st >>= 1) {
    if (tid < st) red[tid] += red[tid + st];
    __syncthreads();
  }
  float r = rsqrtf(red[0] / 128.f + RMS_EPS);
  float gv = gp[(size_t)row * 128 + tid];
  float sw = gv / (1.f + expf(-gv));
  og16[(size_t)row * 128 + tid] = (h16_t)(v * r * onw[tid] * sw);
}

// ---------------------------------------------------------------------------
// m16 = silu(p1) * p3  (f16 out)
// ---------------------------------------------------------------------------
__global__ void k_silu_mul(const float* __restrict__ p1, const float* __restrict__ p3,
                           h16_t* __restrict__ out, long n) {
  long i = (long)blockIdx.x * 256 + threadIdx.x;
  if (i >= n) return;
  float a = p1[i];
  out[i] = (h16_t)(a / (1.f + expf(-a)) * p3[i]);
}

// ---------------------------------------------------------------------------
extern "C" void kernel_launch(void* const* d_in, const int* in_sizes, int n_in,
                              void* d_out, int out_size, void* d_ws, size_t ws_size,
                              hipStream_t stream) {
  (void)in_sizes; (void)n_in; (void)out_size; (void)ws_size;
  const int Bb = 2, Tt = 1024, Dd = 1024, Hh = 16;
  const int Kk = 1024, Vv = 2048, Ii = 2816;
  const int BT = Bb * Tt;

  const float* x        = (const float*)d_in[0];
  const float* norm1_w  = (const float*)d_in[1];
  const float* wq       = (const float*)d_in[2];
  const float* wk       = (const float*)d_in[3];
  const float* wv       = (const float*)d_in[4];
  const float* conv_q   = (const float*)d_in[5];
  const float* conv_k   = (const float*)d_in[6];
  const float* conv_v   = (const float*)d_in[7];
  const float* wb       = (const float*)d_in[8];
  const float* wa       = (const float*)d_in[9];
  const float* A_log    = (const float*)d_in[10];
  const float* dt_bias  = (const float*)d_in[11];
  const float* wg       = (const float*)d_in[12];
  const float* o_norm_w = (const float*)d_in[13];
  const float* wo       = (const float*)d_in[14];
  const float* norm2_w  = (const float*)d_in[15];
  const float* w1       = (const float*)d_in[16];
  const float* w2       = (const float*)d_in[17];
  const float* w3       = (const float*)d_in[18];
  float* out = (float*)d_out;

  // ---- workspace bump allocator (256B aligned) ----
  char* base = (char*)d_ws;
  size_t off = 0;
  auto alloc = [&](size_t bytes) -> void* {
    off = (off + 255) & ~(size_t)255;
    void* p = base + off;
    off += bytes;
    return p;
  };
  h16_t* h16  = (h16_t*)alloc((size_t)BT * Dd * 2);      // reused for h2
  h16_t* wq16 = (h16_t*)alloc((size_t)Dd * Kk * 2);
  h16_t* wk16 = (h16_t*)alloc((size_t)Dd * Kk * 2);
  h16_t* wv16 = (h16_t*)alloc((size_t)Dd * Vv * 2);
  h16_t* wg16 = (h16_t*)alloc((size_t)Dd * Vv * 2);
  h16_t* wo16 = (h16_t*)alloc((size_t)Vv * Dd * 2);
  h16_t* w116 = (h16_t*)alloc((size_t)Dd * Ii * 2);
  h16_t* w316 = (h16_t*)alloc((size_t)Dd * Ii * 2);
  h16_t* w216 = (h16_t*)alloc((size_t)Ii * Dd * 2);
  float* qpre = (float*)alloc((size_t)BT * Kk * 4);      // arena A (contiguous)
  float* kpre = (float*)alloc((size_t)BT * Kk * 4);
  float* vpre = (float*)alloc((size_t)BT * Vv * 4);
  float* qc   = (float*)alloc((size_t)BT * Kk * 4);      // arena B (contiguous)
  float* kc   = (float*)alloc((size_t)BT * Kk * 4);
  float* vc   = (float*)alloc((size_t)BT * Vv * 4);
  float* beta = (float*)alloc((size_t)BT * Hh * 4);
  float* gdec = (float*)alloc((size_t)BT * Hh * 4);
  float* obuf = (float*)alloc((size_t)BT * Vv * 4);
  float* x1   = (float*)alloc((size_t)BT * Dd * 4);
  h16_t* m16  = (h16_t*)alloc((size_t)BT * Ii * 2);
  // phase-based reuse:
  float* gatepre = qpre;          // 16MB over qpre+kpre, after conv consumed them
  h16_t* og16    = (h16_t*)qc;    // after scan consumed qc/kc/vc
  float* p1      = qpre;          // 23MB over arena A, after gate consumed gatepre
  float* p3      = qc;            // 23MB over arena B, after x1 consumed og16

  auto cdiv = [](long a, long b) { return (int)((a + b - 1) / b); };

  // ---- weight conversions (fp32 [K,N] -> f16 [N,K]) ----
  k_cvt_tr<<<cdiv((long)Dd * Kk, 256), 256, 0, stream>>>(wq, wq16, Dd, Kk);
  k_cvt_tr<<<cdiv((long)Dd * Kk, 256), 256, 0, stream>>>(wk, wk16, Dd, Kk);
  k_cvt_tr<<<cdiv((long)Dd * Vv, 256), 256, 0, stream>>>(wv, wv16, Dd, Vv);
  k_cvt_tr<<<cdiv((long)Dd * Vv, 256), 256, 0, stream>>>(wg, wg16, Dd, Vv);
  k_cvt_tr<<<cdiv((long)Vv * Dd, 256), 256, 0, stream>>>(wo, wo16, Vv, Dd);
  k_cvt_tr<<<cdiv((long)Dd * Ii, 256), 256, 0, stream>>>(w1, w116, Dd, Ii);
  k_cvt_tr<<<cdiv((long)Dd * Ii, 256), 256, 0, stream>>>(w3, w316, Dd, Ii);
  k_cvt_tr<<<cdiv((long)Ii * Dd, 256), 256, 0, stream>>>(w2, w216, Ii, Dd);

  // ---- attention half ----
  k_rmsnorm_f16<<<BT, 256, 0, stream>>>(x, norm1_w, h16, Dd);

  dim3 g16x16(BT / 128, Kk / 64), g16x32(BT / 128, Vv / 64), g16x44(BT / 128, Ii / 64);
  k_gemm_wmma<<<g16x16, 256, 0, stream>>>(h16, wq16, qpre, nullptr, BT, Kk, Dd);
  k_gemm_wmma<<<g16x16, 256, 0, stream>>>(h16, wk16, kpre, nullptr, BT, Kk, Dd);
  k_gemm_wmma<<<g16x32, 256, 0, stream>>>(h16, wv16, vpre, nullptr, BT, Vv, Dd);
  k_small_proj<<<BT, 32, 0, stream>>>(h16, wb, wa, A_log, dt_bias, beta, gdec, Dd, Hh);

  k_conv_silu<<<cdiv((long)BT * Kk, 256), 256, 0, stream>>>(qpre, conv_q, qc, Kk, Tt, (long)BT * Kk);
  k_conv_silu<<<cdiv((long)BT * Kk, 256), 256, 0, stream>>>(kpre, conv_k, kc, Kk, Tt, (long)BT * Kk);
  k_conv_silu<<<cdiv((long)BT * Vv, 256), 256, 0, stream>>>(vpre, conv_v, vc, Vv, Tt, (long)BT * Vv);

  k_gemm_wmma<<<g16x32, 256, 0, stream>>>(h16, wg16, gatepre, nullptr, BT, Vv, Dd);

  k_l2norm<<<BT * Kk / 64, 64, 0, stream>>>(qc, 0.125f);  // DK^-0.5 = 1/8
  k_l2norm<<<BT * Kk / 64, 64, 0, stream>>>(kc, 1.0f);

  k_scan<<<Bb * Hh, 128, 0, stream>>>(qc, kc, vc, gdec, beta, obuf, Tt, Hh);
  k_gate<<<BT * Hh, 128, 0, stream>>>(obuf, gatepre, o_norm_w, og16);

  k_gemm_wmma<<<g16x16, 256, 0, stream>>>(og16, wo16, x1, x, BT, Dd, Vv);

  // ---- SwiGLU MLP half ----
  k_rmsnorm_f16<<<BT, 256, 0, stream>>>(x1, norm2_w, h16, Dd);
  k_gemm_wmma<<<g16x44, 256, 0, stream>>>(h16, w116, p1, nullptr, BT, Ii, Dd);
  k_gemm_wmma<<<g16x44, 256, 0, stream>>>(h16, w316, p3, nullptr, BT, Ii, Dd);
  k_silu_mul<<<cdiv((long)BT * Ii, 256), 256, 0, stream>>>(p1, p3, m16, (long)BT * Ii);
  k_gemm_wmma<<<g16x16, 256, 0, stream>>>(m16, w216, out, x1, BT, Dd, Ii);
}
